// ThermodynamicAttention_67894843015412
// MI455X (gfx1250) — compile-verified, matrix operand order checked by reading, not run on hardware
//
#include <hip/hip_runtime.h>

// ---------------------------------------------------------------------------
// ThermodynamicAttention for MI455X (gfx1250, wave32, WMMA bf16)
//   B=4, S=2048, D=1024
//   out = concat( attention_output[4*2048*1024] (f32), entropy_mean[1] (f32) )
//
// ws layout: Q bf16 [b][s][d] | K bf16 [b][s][d] | V bf16 TRANSPOSED [b][d][s]
// ---------------------------------------------------------------------------

#define D_MODEL 1024
#define SEQ     2048
#define BATCH   4

typedef __attribute__((ext_vector_type(16))) __bf16 v16bf;
typedef __attribute__((ext_vector_type(8)))  __bf16 v8bf;
typedef __attribute__((ext_vector_type(8)))  float  v8f;
typedef int vs4i __attribute__((vector_size(16)));   // matches async-LDS builtin

#define AS1 __attribute__((address_space(1)))
#define AS3 __attribute__((address_space(3)))

#if defined(__has_builtin)
#if __has_builtin(__builtin_amdgcn_global_load_async_to_lds_b128)
#define HAVE_ASYNC_LDS 1
#endif
#if __has_builtin(__builtin_amdgcn_s_wait_asynccnt)
#define HAVE_WAIT_ASYNC 1
#endif
#endif

// ---- helpers ---------------------------------------------------------------

// native fp32 -> bf16 (RTE) — let the backend use the hardware convert
__device__ __forceinline__ __bf16 f2bf(float f) { return static_cast<__bf16>(f); }

__device__ __forceinline__ unsigned short f2bfu(float f) {
  union { __bf16 b; unsigned short u; } cv;
  cv.b = static_cast<__bf16>(f);
  return cv.u;
}

__device__ __forceinline__ v16bf combine(v8bf lo, v8bf hi) {
  union { v16bf v; v8bf h[2]; } u;
  u.h[0] = lo; u.h[1] = hi;
  return u.v;
}

// Per-lane fragment = two contiguous 8-element bf16 runs:
//   elements 0..7  -> K = 8g .. 8g+7
//   elements 8..15 -> K = 16+8g .. 23+8g        (g = lane>>4)
// A-fragment (16x32 row-major, K contiguous): row m = lane&15
__device__ __forceinline__ v16bf load_a_bf(const __bf16* __restrict__ src, int ld) {
  int lane = threadIdx.x & 31;
  int m = lane & 15, g = lane >> 4;
  const __bf16* p = src + (size_t)m * ld + 8 * g;
  return combine(*reinterpret_cast<const v8bf*>(p),
                 *reinterpret_cast<const v8bf*>(p + 16));
}

__device__ __forceinline__ v16bf load_a_lds(const unsigned short* __restrict__ src, int ld) {
  int lane = threadIdx.x & 31;
  int m = lane & 15, g = lane >> 4;
  const unsigned short* p = src + (size_t)m * ld + 8 * g;
  return combine(*reinterpret_cast<const v8bf*>(p),
                 *reinterpret_cast<const v8bf*>(p + 16));
}

// B-fragment (32x16) with K contiguous in memory: element (k,n) at src[n*ldn + k]
__device__ __forceinline__ v16bf load_b_bf(const __bf16* __restrict__ src, int ldn) {
  int lane = threadIdx.x & 31;
  int n = lane & 15, g = lane >> 4;
  const __bf16* p = src + (size_t)n * ldn + 8 * g;
  return combine(*reinterpret_cast<const v8bf*>(p),
                 *reinterpret_cast<const v8bf*>(p + 16));
}

__device__ __forceinline__ v16bf load_b_lds(const unsigned short* __restrict__ src, int ldn) {
  int lane = threadIdx.x & 31;
  int n = lane & 15, g = lane >> 4;
  const unsigned short* p = src + (size_t)n * ldn + 8 * g;
  return combine(*reinterpret_cast<const v8bf*>(p),
                 *reinterpret_cast<const v8bf*>(p + 16));
}

// A-fragment from fp32 row-major source (two 8-float runs -> 4x float4 loads)
__device__ __forceinline__ v16bf load_a_f32(const float* __restrict__ src, int ld) {
  int lane = threadIdx.x & 31;
  int m = lane & 15, g = lane >> 4;
  const float* p = src + (size_t)m * ld + 8 * g;
  float4 a0 = reinterpret_cast<const float4*>(p)[0];
  float4 a1 = reinterpret_cast<const float4*>(p)[1];
  float4 b0 = reinterpret_cast<const float4*>(p + 16)[0];
  float4 b1 = reinterpret_cast<const float4*>(p + 16)[1];
  v16bf a;
  a[0] = f2bf(a0.x);  a[1] = f2bf(a0.y);  a[2]  = f2bf(a0.z);  a[3]  = f2bf(a0.w);
  a[4] = f2bf(a1.x);  a[5] = f2bf(a1.y);  a[6]  = f2bf(a1.z);  a[7]  = f2bf(a1.w);
  a[8] = f2bf(b0.x);  a[9] = f2bf(b0.y);  a[10] = f2bf(b0.z);  a[11] = f2bf(b0.w);
  a[12] = f2bf(b1.x); a[13] = f2bf(b1.y); a[14] = f2bf(b1.z);  a[15] = f2bf(b1.w);
  return a;
}

__device__ __forceinline__ v8f wmma_bf16(v16bf a, v16bf b, v8f c) {
  return __builtin_amdgcn_wmma_f32_16x16x32_bf16(
      /*neg_a=*/false, a, /*neg_b=*/false, b,
      /*c_mod=*/(short)0, c, /*reuse_a=*/false, /*reuse_b=*/false);
}

// ---------------------------------------------------------------------------
// Kernel 1: fused QKV projection GEMM, fp32 in -> bf16 out (workspace).
//   Block = 256 threads (8 waves) computes a 128(M) x 64(N) tile; the 64-col
//   W strip is staged transposed into LDS so B-fragments are ds_load_b128.
//   grid.z selects {Q,K,V}; V is stored TRANSPOSED [b][d][s].
// ---------------------------------------------------------------------------
__global__ __launch_bounds__(256) void ta_qkv_kernel(
    const float* __restrict__ x,
    const float* __restrict__ Wq, const float* __restrict__ bq,
    const float* __restrict__ Wk, const float* __restrict__ bk,
    const float* __restrict__ Wv, const float* __restrict__ bv,
    unsigned short* __restrict__ Qo,
    unsigned short* __restrict__ Ko,
    unsigned short* __restrict__ Vo) {
  const int D = D_MODEL, S = SEQ;
  const float* W;
  const float* bias;
  unsigned short* outp;
  if (blockIdx.z == 0)      { W = Wq; bias = bq; outp = Qo; }
  else if (blockIdx.z == 1) { W = Wk; bias = bk; outp = Ko; }
  else                      { W = Wv; bias = bv; outp = Vo; }
  __bf16* out = reinterpret_cast<__bf16*>(outp);

  const int wave = threadIdx.x >> 5;
  const int lane = threadIdx.x & 31;
  const int half = lane >> 4;
  const int mt   = blockIdx.x >> 4;   // 128-row tile (0..63)
  const int ns   = blockIdx.x & 15;   // 64-col strip (0..15)
  const int n0   = ns * 64;

  // W strip staged transposed: [n within strip][k within 32] bf16
  __shared__ __align__(16) unsigned short s_wt[64 * 32];

  v8f acc[4] = {};
  const float* xa = x + ((size_t)mt * 128 + wave * 16) * D;

  // staging indices: thread t handles k = t&31, n-run = (t>>5)*8
  const int sk = threadIdx.x & 31;
  const int sn = (threadIdx.x >> 5) * 8;

  for (int k0 = 0; k0 < D; k0 += 32) {
    __syncthreads();
    {
      const float* wp = W + (size_t)(k0 + sk) * D + n0 + sn;
      float4 w0 = reinterpret_cast<const float4*>(wp)[0];
      float4 w1 = reinterpret_cast<const float4*>(wp)[1];
      s_wt[(sn + 0) * 32 + sk] = f2bfu(w0.x);
      s_wt[(sn + 1) * 32 + sk] = f2bfu(w0.y);
      s_wt[(sn + 2) * 32 + sk] = f2bfu(w0.z);
      s_wt[(sn + 3) * 32 + sk] = f2bfu(w0.w);
      s_wt[(sn + 4) * 32 + sk] = f2bfu(w1.x);
      s_wt[(sn + 5) * 32 + sk] = f2bfu(w1.y);
      s_wt[(sn + 6) * 32 + sk] = f2bfu(w1.z);
      s_wt[(sn + 7) * 32 + sk] = f2bfu(w1.w);
    }
    __syncthreads();

    // batch fragment loads ahead of the WMMA group
    v16bf a = load_a_f32(xa + k0, D);
    v16bf bfr[4];
#pragma unroll
    for (int f = 0; f < 4; ++f) bfr[f] = load_b_lds(s_wt + f * 16 * 32, 32);
#pragma unroll
    for (int f = 0; f < 4; ++f) acc[f] = wmma_bf16(a, bfr[f], acc[f]);
  }

  // epilogue: + bias, convert to bf16, store (V transposed)
#pragma unroll
  for (int f = 0; f < 4; ++f) {
    int col = n0 + f * 16 + (lane & 15);
    float bcol = bias[col];
#pragma unroll
    for (int r = 0; r < 8; ++r) {
      int row = mt * 128 + wave * 16 + r + 8 * half;   // global row over B*S
      __bf16 val = f2bf(acc[f][r] + bcol);
      if (blockIdx.z == 2) {
        int bb = row >> 11, s = row & (SEQ - 1);
        out[((size_t)bb * D + col) * S + s] = val;     // [b][d][s]
      } else {
        out[(size_t)row * D + col] = val;              // [b][s][d]
      }
    }
  }
}

// ---------------------------------------------------------------------------
// Kernel 2: flash-style attention + entropy.
//   One workgroup (8 waves) per (batch, 16-row query tile).
//   Wave w owns output dims [w*128, (w+1)*128).  Q tile staged in LDS
//   via GLOBAL_LOAD_ASYNC_TO_LDS_B128 when the builtin is available.
// ---------------------------------------------------------------------------
__global__ __launch_bounds__(256) void ta_attn_kernel(
    const unsigned short* __restrict__ Qw,
    const unsigned short* __restrict__ Kw,
    const unsigned short* __restrict__ Vw,   // transposed [b][d][s]
    const float* __restrict__ temperature,
    float* __restrict__ out,
    float* __restrict__ entropy_out) {
  const int D = D_MODEL, S = SEQ;
  const int b    = blockIdx.y;
  const int qt   = blockIdx.x;            // query tile 0..127
  const int wave = threadIdx.x >> 5;      // 0..7
  const int lane = threadIdx.x & 31;
  const int row  = lane & 15;             // row this lane owns for softmax stats
  const int half = lane >> 4;

  const __bf16* Qb  = reinterpret_cast<const __bf16*>(Qw) + (size_t)b * S * D + (size_t)qt * 16 * D;
  const __bf16* Kb  = reinterpret_cast<const __bf16*>(Kw) + (size_t)b * S * D;
  const __bf16* Vbt = reinterpret_cast<const __bf16*>(Vw) + (size_t)b * D * S;

  const float scale = 0.125f / temperature[0];   // 1/sqrt(HEAD_DIM=64) / temp

  __shared__ __align__(16) unsigned short s_q[16 * 1024];      // 32 KB Q tile
  __shared__ __align__(16) float          s_scores[16 * 128];  //  8 KB
  __shared__ __align__(16) unsigned short s_probs[16 * 128];   //  4 KB

  // stage the 16x1024 Q tile (contiguous, 32 KB) into LDS
#if defined(HAVE_ASYNC_LDS)
  {
    AS1 vs4i* g = (AS1 vs4i*)(void*)const_cast<__bf16*>(Qb);
    AS3 vs4i* l = (AS3 vs4i*)(void*)s_q;
#pragma unroll
    for (int i = 0; i < 8; ++i) {
      int idx = threadIdx.x + i * 256;              // 2048 x 16B chunks
      __builtin_amdgcn_global_load_async_to_lds_b128(g + idx, l + idx, 0, 0);
    }
#if defined(HAVE_WAIT_ASYNC)
    __builtin_amdgcn_s_wait_asynccnt(0);
#else
    asm volatile("s_wait_asynccnt 0x0" ::: "memory");
#endif
  }
#else
  {
    const v8bf* src = reinterpret_cast<const v8bf*>(Qb);
    v8bf* dst = reinterpret_cast<v8bf*>(s_q);
    for (int i = threadIdx.x; i < 16 * 1024 / 8; i += 256) dst[i] = src[i];
  }
#endif
  __syncthreads();

  v8f acc[8] = {};
  float m = -1e30f, l = 0.0f, t = 0.0f;          // online softmax + entropy state

  for (int kb = 0; kb < S; kb += 128) {
    if (kb + 128 < S) {                          // global_prefetch_b8 of next block
      __builtin_prefetch(Kb + (size_t)(kb + 128) * D, 0, 1);
      __builtin_prefetch(Vbt + (size_t)(wave * 128) * S + kb + 128, 0, 1);
    }
    // ---- phase 1: this wave's 16x16 score tile for keys [kb+wave*16, +16) ----
    v8f sc = {};
    const __bf16* Kt = Kb + (size_t)(kb + wave * 16) * D;
#pragma unroll 2
    for (int k0 = 0; k0 < D; k0 += 32) {
      v16bf a  = load_a_lds(s_q + k0, 1024);
      v16bf bf = load_b_bf(Kt + k0, D);          // (k=d, n=key): Kt[n*D + k]
      sc = wmma_bf16(a, bf, sc);
    }
#pragma unroll
    for (int r = 0; r < 8; ++r) {
      int mr = r + 8 * half;
      s_scores[mr * 128 + wave * 16 + (lane & 15)] = sc[r] * scale;
    }
    __syncthreads();

    // ---- phase 2: per-row online softmax stats (redundant per wave) ----
    const float4* srow = reinterpret_cast<const float4*>(s_scores + row * 128 + half * 64);
    float bm = -1e30f;
#pragma unroll
    for (int c = 0; c < 16; ++c) {
      float4 v = srow[c];
      bm = fmaxf(bm, fmaxf(fmaxf(v.x, v.y), fmaxf(v.z, v.w)));
    }
    bm = fmaxf(bm, __shfl_xor(bm, 16, 32));
    float mn    = fmaxf(m, bm);
    float alpha = __expf(m - mn);
    float sblk = 0.0f, tblk = 0.0f;
#pragma unroll
    for (int c = 0; c < 16; ++c) {
      float4 v = srow[c];
      float e0 = __expf(v.x - mn), e1 = __expf(v.y - mn);
      float e2 = __expf(v.z - mn), e3 = __expf(v.w - mn);
      sblk += (e0 + e1) + (e2 + e3);
      tblk += e0 * (v.x - mn) + e1 * (v.y - mn) + e2 * (v.z - mn) + e3 * (v.w - mn);
    }
    sblk += __shfl_xor(sblk, 16, 32);
    tblk += __shfl_xor(tblk, 16, 32);
    t = alpha * (t + (m - mn) * l) + tblk;
    l = alpha * l + sblk;
    m = mn;
    // this wave writes unnormalized probs for columns [wave*16, wave*16+16)
#pragma unroll
    for (int c = 0; c < 8; ++c) {
      int col = wave * 16 + half * 8 + c;
      s_probs[row * 128 + col] = f2bfu(__expf(s_scores[row * 128 + col] - mn));
    }
    __syncthreads();

    // ---- phase 3: rescale accumulators, accumulate P(16x128) @ V slice ----
    float al[8];
#pragma unroll
    for (int r = 0; r < 8; ++r) al[r] = __shfl(alpha, r + 8 * half, 32);
#pragma unroll
    for (int f = 0; f < 8; ++f)
#pragma unroll
      for (int r = 0; r < 8; ++r) acc[f][r] *= al[r];

#pragma unroll
    for (int kc = 0; kc < 4; ++kc) {
      v16bf a = load_a_lds(s_probs + kc * 32, 128);
      // V transposed: element (k=key, n=dim) at Vbt[(dim)*S + key]
      const __bf16* Vt = Vbt + (size_t)(wave * 128) * S + kb + kc * 32;
#pragma unroll
      for (int fg = 0; fg < 2; ++fg) {
        v16bf bfr[4];
#pragma unroll
        for (int f = 0; f < 4; ++f)
          bfr[f] = load_b_bf(Vt + (size_t)((fg * 4 + f) * 16) * S, S);
#pragma unroll
        for (int f = 0; f < 4; ++f)
          acc[fg * 4 + f] = wmma_bf16(a, bfr[f], acc[fg * 4 + f]);
      }
    }
    __syncthreads();
  }

  // ---- epilogue: normalize by 1/l, store fp32 output ----
  float invl = 1.0f / l;
  float* Ob = out + (size_t)b * S * D + (size_t)qt * 16 * D;
  float il[8];
#pragma unroll
  for (int r = 0; r < 8; ++r) il[r] = __shfl(invl, r + 8 * half, 32);
#pragma unroll
  for (int f = 0; f < 8; ++f)
#pragma unroll
    for (int r = 0; r < 8; ++r) {
      int mr = r + 8 * half;
      Ob[(size_t)mr * D + wave * 128 + f * 16 + (lane & 15)] = acc[f][r] * il[r];
    }

  // ---- entropy: H_row = log l - t/l ; mean over B*S rows ----
  if (wave == 0 && lane < 16) {
    float ent = __logf(l) - t * invl;
    atomicAdd(entropy_out, ent * (1.0f / (float)(BATCH * SEQ)));
  }
}

// ---------------------------------------------------------------------------
// Launch
// ---------------------------------------------------------------------------
extern "C" void kernel_launch(void* const* d_in, const int* in_sizes, int n_in,
                              void* d_out, int out_size, void* d_ws, size_t ws_size,
                              hipStream_t stream) {
  const float* x    = (const float*)d_in[0];
  const float* Wq   = (const float*)d_in[1];
  const float* bq   = (const float*)d_in[2];
  const float* Wk   = (const float*)d_in[3];
  const float* bk   = (const float*)d_in[4];
  const float* Wv   = (const float*)d_in[5];
  const float* bv   = (const float*)d_in[6];
  const float* temp = (const float*)d_in[7];
  float* out = (float*)d_out;

  const size_t NE = (size_t)BATCH * SEQ * D_MODEL;   // 8,388,608 elements
  unsigned short* Qw = (unsigned short*)d_ws;        // bf16 bits, 16 MB each
  unsigned short* Kw = Qw + NE;
  unsigned short* Vw = Kw + NE;                      // transposed [b][d][s]

  // zero the entropy accumulator slot (last output element)
  (void)hipMemsetAsync(out + NE, 0, sizeof(float), stream);

  // QKV projection: 64 M-tiles(128 rows) x 16 N-strips(64 cols), z = q/k/v
  dim3 g1(1024, 1, 3);
  ta_qkv_kernel<<<g1, 256, 0, stream>>>(x, Wq, bq, Wk, bk, Wv, bv, Qw, Kw, Vw);

  // Attention: (query tile, batch), 8 waves per block
  dim3 g2(SEQ / 16, BATCH, 1);
  ta_attn_kernel<<<g2, 256, 0, stream>>>(Qw, Kw, Vw, temp, out, out + NE);
}
